// BoxHead_2138893714091
// MI455X (gfx1250) — matrix operand order repaired and checked by reading, not compiled, not run on hardware
//
#include <hip/hip_runtime.h>

// ---------- types ----------
typedef __attribute__((ext_vector_type(16))) __bf16         v16bf;
typedef __attribute__((ext_vector_type(8)))  float          v8f;
typedef __attribute__((ext_vector_type(4)))  unsigned int   v4u;
typedef __attribute__((ext_vector_type(8)))  unsigned int   v8u;
typedef __attribute__((ext_vector_type(4)))  unsigned short v4h;
typedef unsigned short u16;

// fp32 -> bf16, round-to-nearest-even
__device__ __forceinline__ u16 f2bf(float f) {
  union { float f; unsigned u; } c; c.f = f;
  unsigned r = c.u + 0x7FFFu + ((c.u >> 16) & 1u);
  return (u16)(r >> 16);
}

// assemble a 16-element bf16 WMMA fragment from two 16B chunks
__device__ __forceinline__ v16bf mkfrag(v4u lo, v4u hi) {
  union { v8u u; v16bf b; } cv;
  cv.u[0] = lo[0]; cv.u[1] = lo[1]; cv.u[2] = lo[2]; cv.u[3] = lo[3];
  cv.u[4] = hi[0]; cv.u[5] = hi[1]; cv.u[6] = hi[2]; cv.u[7] = hi[3];
  return cv.b;
}

// ---------- CDNA5 async / tensor data movement ----------
// Async global->LDS copy (ASYNCcnt). LDS address = low 32 bits of flat address
// (ISA 10.2 aperture rule).
__device__ __forceinline__ void async_b128(unsigned lds, const void* g) {
  asm volatile("global_load_async_to_lds_b128 %0, %1, off"
               :: "v"(lds), "v"(g) : "memory");
}
__device__ __forceinline__ void wait_async_le10() {
  asm volatile("s_wait_asynccnt 0xa" ::: "memory");
}
__device__ __forceinline__ void wait_async_le0() {
  asm volatile("s_wait_asynccnt 0x0" ::: "memory");
}
// TDM tensor load, 2-group descriptor form (2D tile), TENSORcnt-tracked.
__device__ __forceinline__ void tdm_load_2g(v4u g0, v8u g1) {
  asm volatile("tensor_load_to_lds %0, %1" :: "s"(g0), "s"(g1) : "memory");
}

// ---------- elementwise fp32 -> bf16 (vectorized) ----------
__global__ void k_cvt_bf16(const float* __restrict__ in, u16* __restrict__ out, long n4) {
  long i = (long)blockIdx.x * blockDim.x + threadIdx.x;
  long stride = (long)gridDim.x * blockDim.x;
  for (; i < n4; i += stride) {
    float4 v = ((const float4*)in)[i];
    v4h o = { f2bf(v.x), f2bf(v.y), f2bf(v.z), f2bf(v.w) };
    ((v4h*)out)[i] = o;
  }
}

// ---------- tiled transpose: in[R,C] fp32 -> out[C,R] bf16 ----------
__global__ __launch_bounds__(256)
void k_transpose_bf16(const float* __restrict__ in, u16* __restrict__ out, int R, int C) {
  __shared__ u16 tile[32][33];
  const int cb = blockIdx.x * 32, rb = blockIdx.y * 32;
  const int tx = threadIdx.x & 31, ty = threadIdx.x >> 5;  // 32 x 8
  #pragma unroll
  for (int i = 0; i < 32; i += 8) {
    int r = rb + ty + i, c = cb + tx;
    if (r < R && c < C) tile[ty + i][tx] = f2bf(in[(size_t)r * C + c]);
  }
  __syncthreads();
  #pragma unroll
  for (int i = 0; i < 32; i += 8) {
    int r2 = cb + ty + i, c2 = rb + tx;   // out is [C,R]
    if (r2 < C && c2 < R) out[(size_t)r2 * R + c2] = tile[tx][ty + i];
  }
}

// ---------- build fused heads weights: WcrT[16,1024] bf16, biasC[16] ----------
__global__ void k_build_heads(const float* __restrict__ Wc, const float* __restrict__ Wr,
                              const float* __restrict__ bc, const float* __restrict__ br,
                              u16* __restrict__ WcrT, float* __restrict__ biasC) {
  int i = blockIdx.x * blockDim.x + threadIdx.x;
  if (i < 16 * 1024) {
    int j = i >> 10, k = i & 1023;
    float v = (j < 4) ? Wc[k * 4 + j] : Wr[k * 12 + (j - 4)];
    WcrT[i] = f2bf(v);
  }
  if (i < 16) biasC[i] = (i < 4) ? bc[i] : br[i - 4];
}

// ===========================================================================
// Async double-buffered GEMM (A bf16): H = relu(A[M,K] * Bt[N,K]^T + bias)
// BM=64 BN=256 BK=64; 8 waves (2x4); wave tile 32x64; K % 64 == 0.
// Tiles move via global_load_async_to_lds_b128 + s_wait_asynccnt pipelining.
// ===========================================================================
__global__ __launch_bounds__(256, 1)
void k_gemm_async(const u16* __restrict__ A, const u16* __restrict__ Bt,
                  const float* __restrict__ bias, u16* __restrict__ Hout,
                  int M, int N, int K) {
  constexpr int BM_ = 64, BN_ = 256, BK_ = 64, LDS_ = BK_ + 8;
  __shared__ u16 As[2][BM_ * LDS_];
  __shared__ u16 Bs[2][BN_ * LDS_];

  const int tid  = threadIdx.x;
  const int lane = tid & 31;
  const int wave = tid >> 5;
  const int wr   = (wave >> 2) * 32;
  const int wc   = (wave & 3) * 64;
  const int bRow = blockIdx.y * BM_;
  const int bCol = blockIdx.x * BN_;
  const int mlo  = lane & 15, hsel = lane >> 4;

  v8f acc[2][4];
  const v8f vz = {0.f, 0.f, 0.f, 0.f, 0.f, 0.f, 0.f, 0.f};
  #pragma unroll
  for (int i = 0; i < 2; ++i)
    #pragma unroll
    for (int j = 0; j < 4; ++j) acc[i][j] = vz;

  // staging geometry: each thread moves 2 A chunks + 8 B chunks of 16B per tile
  const int r0 = tid >> 3;          // 0..31
  const int c8 = (tid & 7) << 3;    // element column
  int ga0 = bRow + r0;      ga0 = ga0 < M ? ga0 : M - 1;   // clamp (stores guarded)
  int ga1 = bRow + r0 + 32; ga1 = ga1 < M ? ga1 : M - 1;
  const u16* aP0 = A + (size_t)ga0 * K + c8;
  const u16* aP1 = A + (size_t)ga1 * K + c8;
  const u16* bP  = Bt + (size_t)(bCol + r0) * K + c8;

  const int nT = K / BK_;
  auto issue = [&](int t, int buf) {
    const int k0 = t * BK_;
    async_b128((unsigned)(size_t)&As[buf][r0 * LDS_ + c8],        aP0 + k0);
    async_b128((unsigned)(size_t)&As[buf][(r0 + 32) * LDS_ + c8], aP1 + k0);
    #pragma unroll
    for (int i = 0; i < 8; ++i) {
      async_b128((unsigned)(size_t)&Bs[buf][(r0 + i * 32) * LDS_ + c8],
                 bP + (size_t)(i * 32) * K + k0);
    }
  };

  issue(0, 0);
  for (int t = 0; t < nT; ++t) {
    const int buf = t & 1;
    if (t + 1 < nT) { issue(t + 1, buf ^ 1); wait_async_le10(); }
    else            { wait_async_le0(); }
    __syncthreads();   // tile t visible to all waves

    #pragma unroll
    for (int ks = 0; ks < 2; ++ks) {
      v16bf af[2], bfr[4];
      #pragma unroll
      for (int i = 0; i < 2; ++i) {
        const u16* pa = &As[buf][(wr + i * 16 + mlo) * LDS_ + ks * 32];
        v4u lo = *(const v4u*)(pa + hsel * 8);
        v4u hi = *(const v4u*)(pa + 16 + hsel * 8);
        af[i] = mkfrag(lo, hi);
      }
      #pragma unroll
      for (int j = 0; j < 4; ++j) {
        const u16* pb = &Bs[buf][(wc + j * 16 + mlo) * LDS_ + ks * 32 + hsel * 8];
        v4u lo = *(const v4u*)(pb);
        v4u hi = *(const v4u*)(pb + 16);
        bfr[j] = mkfrag(lo, hi);
      }
      #pragma unroll
      for (int i = 0; i < 2; ++i)
        #pragma unroll
        for (int j = 0; j < 4; ++j)
          acc[i][j] = __builtin_amdgcn_wmma_f32_16x16x32_bf16(
              false, af[i], false, bfr[j], (short)0, acc[i][j], false, false);
    }
    __syncthreads();   // all waves done reading buf before it is async-overwritten
  }

  // epilogue: bias + ReLU + bf16 store
  #pragma unroll
  for (int j = 0; j < 4; ++j) {
    const int n = bCol + wc + j * 16 + mlo;
    const float bv = bias[n];
    #pragma unroll
    for (int i = 0; i < 2; ++i) {
      const int rb = bRow + wr + i * 16 + hsel * 8;
      v8f a = acc[i][j];
      #pragma unroll
      for (int r = 0; r < 8; ++r) {
        int row = rb + r;
        if (row < M) {
          float v = a[r] + bv;
          v = v > 0.f ? v : 0.f;
          Hout[(size_t)row * N + n] = f2bf(v);
        }
      }
    }
  }
}

// ===========================================================================
// Fallback GEMM with fp32 A (on-the-fly bf16 convert) — B tile moved by the
// Tensor Data Mover. TDM pad (interval=32 DW, amount=4 DW) reproduces the
// 72-element LDS row stride used for conflict-free fragment reads.
// ===========================================================================
__global__ __launch_bounds__(256, 1)
void k_gemm_f32A(const float* __restrict__ A, const u16* __restrict__ Bt,
                 const float* __restrict__ bias, u16* __restrict__ Hout,
                 int M, int N, int K) {
  constexpr int BM_ = 64, BN_ = 256, BK_ = 64, LDS_ = BK_ + 8;
  __shared__ u16 As[BM_ * LDS_];
  __shared__ u16 Bs[BN_ * LDS_];

  const int tid  = threadIdx.x;
  const int lane = tid & 31;
  const int wave = tid >> 5;
  const int wr   = (wave >> 2) * 32;
  const int wc   = (wave & 3) * 64;
  const int bRow = blockIdx.y * BM_;
  const int bCol = blockIdx.x * BN_;
  const int mlo  = lane & 15, hsel = lane >> 4;

  v8f acc[2][4];
  const v8f vz = {0.f, 0.f, 0.f, 0.f, 0.f, 0.f, 0.f, 0.f};
  #pragma unroll
  for (int i = 0; i < 2; ++i)
    #pragma unroll
    for (int j = 0; j < 4; ++j) acc[i][j] = vz;

  // TDM descriptor group 1 (constant across K loop):
  // data_size=2B, pad_enable, pad_interval=32 DWORDs, pad_amount=4 DWORDs,
  // tensor_dim0=K, tensor_dim1=N, tile_dim0=64, tile_dim1=256, stride0=K.
  v8u g1;
  g1[0] = (1u << 16) | (1u << 20) | (4u << 22) | (3u << 25);
  g1[1] = ((unsigned)K & 0xFFFFu) << 16;
  g1[2] = ((unsigned)K >> 16) | (((unsigned)N & 0xFFFFu) << 16);
  g1[3] = ((unsigned)N >> 16) | (64u << 16);
  g1[4] = 256u;
  g1[5] = (unsigned)K;
  g1[6] = 0u;
  g1[7] = 0u;
  const unsigned ldsB = (unsigned)(size_t)&Bs[0];

  for (int k0 = 0; k0 < K; k0 += BK_) {
    // stage A tile: fp32 -> bf16 in registers -> LDS
    {
      const int r0 = tid >> 4;          // 0..15
      const int c4 = (tid & 15) << 2;   // 0..60
      #pragma unroll
      for (int i = 0; i < 4; ++i) {
        int r = r0 + i * 16;
        int gr = bRow + r; gr = gr < M ? gr : M - 1;
        float4 v = *(const float4*)(A + (size_t)gr * K + k0 + c4);
        v4h o = { f2bf(v.x), f2bf(v.y), f2bf(v.z), f2bf(v.w) };
        *(v4h*)&As[r * LDS_ + c4] = o;
      }
      if (k0 + BK_ < K) {
        int pr = bRow + r0; pr = pr < M ? pr : M - 1;
        __builtin_prefetch(A + (size_t)pr * K + k0 + BK_, 0, 0);
      }
    }
    // stage B tile via Tensor Data Mover (one descriptor, wave 0 only)
    if (wave == 0) {
      unsigned long long ga = (unsigned long long)(size_t)(Bt + (size_t)bCol * K + k0);
      v4u g0;
      g0[0] = 1u;                                    // count=1, user descriptor
      g0[1] = ldsB;                                  // lds_addr
      g0[2] = (unsigned)ga;                          // global_addr[31:0]
      g0[3] = (unsigned)(ga >> 32) | 0x80000000u;    // global_addr[56:32] | type=2
      tdm_load_2g(g0, g1);
      __builtin_amdgcn_s_wait_tensorcnt(0);
    }
    __syncthreads();

    #pragma unroll
    for (int ks = 0; ks < 2; ++ks) {
      v16bf af[2], bfr[4];
      #pragma unroll
      for (int i = 0; i < 2; ++i) {
        const u16* pa = &As[(wr + i * 16 + mlo) * LDS_ + ks * 32];
        v4u lo = *(const v4u*)(pa + hsel * 8);
        v4u hi = *(const v4u*)(pa + 16 + hsel * 8);
        af[i] = mkfrag(lo, hi);
      }
      #pragma unroll
      for (int j = 0; j < 4; ++j) {
        const u16* pb = &Bs[(wc + j * 16 + mlo) * LDS_ + ks * 32 + hsel * 8];
        v4u lo = *(const v4u*)(pb);
        v4u hi = *(const v4u*)(pb + 16);
        bfr[j] = mkfrag(lo, hi);
      }
      #pragma unroll
      for (int i = 0; i < 2; ++i)
        #pragma unroll
        for (int j = 0; j < 4; ++j)
          acc[i][j] = __builtin_amdgcn_wmma_f32_16x16x32_bf16(
              false, af[i], false, bfr[j], (short)0, acc[i][j], false, false);
    }
    __syncthreads();
  }

  #pragma unroll
  for (int j = 0; j < 4; ++j) {
    const int n = bCol + wc + j * 16 + mlo;
    const float bv = bias[n];
    #pragma unroll
    for (int i = 0; i < 2; ++i) {
      const int rb = bRow + wr + i * 16 + hsel * 8;
      v8f a = acc[i][j];
      #pragma unroll
      for (int r = 0; r < 8; ++r) {
        int row = rb + r;
        if (row < M) {
          float v = a[r] + bv;
          v = v > 0.f ? v : 0.f;
          Hout[(size_t)row * N + n] = f2bf(v);
        }
      }
    }
  }
}

// ---------- heads: 16 rows x 16 fused cols per wave over K=1024 ----------
__global__ __launch_bounds__(32)
void k_heads(const u16* __restrict__ H, const u16* __restrict__ WcrT,
             const float* __restrict__ biasC, float* __restrict__ logits,
             float* __restrict__ box, int M) {
  const int lane = threadIdx.x & 31;
  const int row0 = blockIdx.x * 16;
  const int nm = lane & 15, h = lane >> 4;
  v8f acc = {0.f, 0.f, 0.f, 0.f, 0.f, 0.f, 0.f, 0.f};
  int ar = row0 + nm; ar = ar < M ? ar : M - 1;
  const u16* arow = H + (size_t)ar * 1024;
  const u16* brow = WcrT + (size_t)nm * 1024;
  for (int k0 = 0; k0 < 1024; k0 += 32) {
    v4u alo = *(const v4u*)(arow + k0 + h * 8);
    v4u ahi = *(const v4u*)(arow + k0 + 16 + h * 8);
    v4u blo = *(const v4u*)(brow + k0 + h * 8);
    v4u bhi = *(const v4u*)(brow + k0 + 16 + h * 8);
    acc = __builtin_amdgcn_wmma_f32_16x16x32_bf16(
        false, mkfrag(alo, ahi), false, mkfrag(blo, bhi), (short)0, acc, false, false);
  }
  const float bv = biasC[nm];
  #pragma unroll
  for (int r = 0; r < 8; ++r) {
    int row = row0 + h * 8 + r;
    if (row < M) {
      float v = acc[r] + bv;
      if (nm < 4) logits[(size_t)row * 4 + nm] = v;
      else        box[(size_t)row * 12 + (nm - 4)] = v;
    }
  }
}

// ---------- host ----------
extern "C" void kernel_launch(void* const* d_in, const int* in_sizes, int n_in,
                              void* d_out, int out_size, void* d_ws, size_t ws_size,
                              hipStream_t stream) {
  const float* X  = (const float*)d_in[0];
  const float* W1 = (const float*)d_in[1];
  const float* b1 = (const float*)d_in[2];
  const float* W2 = (const float*)d_in[3];
  const float* b2 = (const float*)d_in[4];
  const float* Wc = (const float*)d_in[5];
  const float* bc = (const float*)d_in[6];
  const float* Wr = (const float*)d_in[7];
  const float* br = (const float*)d_in[8];

  const int KIN = 50176, DH = 1024;
  const int M = in_sizes[0] / KIN;   // 2000

  // workspace partition (256B aligned)
  char* p = (char*)d_ws;
  auto take = [&](size_t bytes) -> void* {
    void* r = (void*)p; p += (bytes + 255) & ~(size_t)255; return r;
  };
  u16*   W1T  = (u16*)  take((size_t)2 * KIN * DH);   // [1024, 50176] bf16 (fits L2)
  u16*   W2T  = (u16*)  take((size_t)2 * DH * DH);    // [1024, 1024]  bf16
  u16*   WcrT = (u16*)  take((size_t)2 * 16 * DH);    // [16, 1024]    bf16
  float* bC   = (float*)take((size_t)64 * 4);
  u16*   h1   = (u16*)  take((size_t)2 * M * DH);
  u16*   h2   = (u16*)  take((size_t)2 * M * DH);
  size_t used = (size_t)(p - (char*)d_ws);
  size_t xbf_bytes = (size_t)2 * M * KIN;
  u16* Xbf = nullptr;
  if (used + xbf_bytes + 256 <= ws_size) Xbf = (u16*)take(xbf_bytes);

  // weight prep
  {
    dim3 g1((DH + 31) / 32, (KIN + 31) / 32);
    k_transpose_bf16<<<g1, 256, 0, stream>>>(W1, W1T, KIN, DH);
    dim3 g2((DH + 31) / 32, (DH + 31) / 32);
    k_transpose_bf16<<<g2, 256, 0, stream>>>(W2, W2T, DH, DH);
    k_build_heads<<<64, 256, 0, stream>>>(Wc, Wr, bc, br, WcrT, bC);
  }

  dim3 gg(DH / 256, (M + 63) / 64);   // (4, 32)
  if (Xbf) {
    long n4 = (long)M * KIN / 4;
    k_cvt_bf16<<<4096, 256, 0, stream>>>(X, Xbf, n4);
    k_gemm_async<<<gg, 256, 0, stream>>>(Xbf, W1T, b1, h1, M, DH, KIN);
  } else {
    k_gemm_f32A<<<gg, 256, 0, stream>>>(X, W1T, b1, h1, M, DH, KIN);
  }
  k_gemm_async<<<gg, 256, 0, stream>>>(h1, W2T, b2, h2, M, DH, DH);

  float* out = (float*)d_out;
  k_heads<<<(M + 15) / 16, 32, 0, stream>>>(h2, WcrT, bC, out, out + (size_t)M * 4, M);
}
